// RitsModel_i_15985868276390
// MI455X (gfx1250) — matrix operand order repaired and checked by reading, not compiled, not run on hardware
//
#include <hip/hip_runtime.h>
#include <hip/hip_bf16.h>
#include <math.h>
#include <stdint.h>

// Problem dims (fixed by reference)
#define B_TOT 512
#define T_LEN 128
#define D_IN  128
#define H_DIM 512
#define BM    16      // batch rows per workgroup -> 32 workgroups

typedef __attribute__((ext_vector_type(16))) __bf16 v16bf;
typedef __attribute__((ext_vector_type(8)))  __bf16 v8bf;
typedef __attribute__((ext_vector_type(8)))  float  v8f;

#define WMMA_BF16(acc, a, b) \
  __builtin_amdgcn_wmma_f32_16x16x32_bf16(false, (a), false, (b), (short)0, (acc), false, false)

__device__ __forceinline__ v8f zero8() {
  v8f z = {0.f,0.f,0.f,0.f,0.f,0.f,0.f,0.f};
  return z;
}

// Launder a (uniform) pointer so loop-invariant weight loads are NOT hoisted
// out of the time loop and spilled to scratch; they must stay as per-iteration
// global_load_b128 that hit the 192MB L2 (weights are L2-resident, shared by
// all workgroups).
template <typename T>
__device__ __forceinline__ const T* opaque_ptr(const T* p) {
  asm volatile("" : "+s"(p));
  return p;
}

// A-fragment (16x32 bf16, row-major activations in LDS).
// lane<16 : row m, K = k0..k0+15
// lane>=16: row m, K = k0+16..k0+31
__device__ __forceinline__ v16bf load_A_frag(const __bf16* base, int lda,
                                             int m, int k0, int lhalf) {
  union { v16bf v; v8bf h[2]; } u;
  const __bf16* p = base + m * lda + k0 + lhalf * 16;
  u.h[0] = *(const v8bf*)(p);
  u.h[1] = *(const v8bf*)(p + 8);
  return u.v;
}

// B-fragment (32x16 bf16) for Y = X @ W^T : B[k][n] = W[n][k], W row-major [N][K].
// lane<16 : col n, K = k0..k0+15 ; lane>=16: col n, K = k0+16..k0+31  (contiguous 32B)
// Dereference through an explicit global (addrspace 1) pointer so the laundered
// base still selects global_load_b128 (not flat_load, which also burns DScnt).
__device__ __forceinline__ v16bf load_B_frag(const __bf16* W, int ldw,
                                             int nrow, int k0, int lhalf) {
  const __bf16* p = W + (size_t)nrow * ldw + k0 + lhalf * 16;
  typedef const __attribute__((address_space(1))) v16bf GV;
  return *reinterpret_cast<GV*>((uintptr_t)p);
}

__device__ __forceinline__ float fast_sigmoid(float x) {
  return 1.0f / (1.0f + __expf(-x));
}

// Software-pipelined 16x16 tile GEMM: A from LDS, B from global (L2-resident).
// Loads for step ks+1 are issued before the WMMA of step ks so the scheduler
// can overlap memory latency with matrix math (partial counted waits).
template <int KS>
__device__ __forceinline__ v8f gemm_pipe(const __bf16* A, int lda,
                                         const __bf16* W, int ldw, int nrow,
                                         int lmod, int lhalf, v8f acc) {
  v16bf a0 = load_A_frag(A, lda, lmod, 0, lhalf);
  v16bf b0 = load_B_frag(W, ldw, nrow, 0, lhalf);
#pragma unroll
  for (int ks = 0; ks < KS; ++ks) {
    v16bf a1, b1;
    if (ks + 1 < KS) {
      a1 = load_A_frag(A, lda, lmod, (ks + 1) * 32, lhalf);
      b1 = load_B_frag(W, ldw, nrow, (ks + 1) * 32, lhalf);
    }
    acc = WMMA_BF16(acc, a0, b0);
    a0 = a1; b0 = b1;
  }
  return acc;
}

// One pipeline stage of the fused gates GEMM: 1 A-fragment + 4 B-fragments
// (i,f,g,o rows of the weight matrix). ks 0..7 -> [c_c|m] @ W_ih (K=256),
// ks 8..23 -> h @ W_hh (K=512). ks is a compile-time constant after unroll.
struct FStep { v16bf a, b0, b1, b2, b3; };

__device__ __forceinline__ void load_fstep(FStep& s, int ks,
                                           const __bf16* cat, const __bf16* hcur,
                                           const __bf16* wih, const __bf16* whh,
                                           int nt, int lmod, int lhalf) {
  if (ks < 8) {
    int k0 = ks * 32;
    int nr = nt * 16 + lmod;
    s.a  = load_A_frag(cat, 256, lmod, k0, lhalf);
    s.b0 = load_B_frag(wih, 256, 0 * H_DIM + nr, k0, lhalf);
    s.b1 = load_B_frag(wih, 256, 1 * H_DIM + nr, k0, lhalf);
    s.b2 = load_B_frag(wih, 256, 2 * H_DIM + nr, k0, lhalf);
    s.b3 = load_B_frag(wih, 256, 3 * H_DIM + nr, k0, lhalf);
  } else {
    int k0 = (ks - 8) * 32;
    int nr = nt * 16 + lmod;
    s.a  = load_A_frag(hcur, 512, lmod, k0, lhalf);
    s.b0 = load_B_frag(whh, 512, 0 * H_DIM + nr, k0, lhalf);
    s.b1 = load_B_frag(whh, 512, 1 * H_DIM + nr, k0, lhalf);
    s.b2 = load_B_frag(whh, 512, 2 * H_DIM + nr, k0, lhalf);
    s.b3 = load_B_frag(whh, 512, 3 * H_DIM + nr, k0, lhalf);
  }
}

// ---------------- prep kernels ----------------

__global__ void k_zero_loss(float* loss) {
  if (threadIdx.x == 0 && blockIdx.x == 0) *loss = 0.f;
}

// msum[t] = sum over (b,d) of masks[b,t,d] + 1e-5  (input-only -> precomputable)
__global__ void k_msum(const float* __restrict__ masks, float* __restrict__ msum) {
  int t = blockIdx.x;
  __shared__ float red[256];
  float s = 0.f;
  for (int i = threadIdx.x; i < B_TOT * D_IN; i += 256) {
    int b = i >> 7, d = i & 127;
    s += masks[(size_t)b * T_LEN * D_IN + (size_t)t * D_IN + d];
  }
  red[threadIdx.x] = s;
  __syncthreads();
  if (threadIdx.x == 0) {
    float tot = 1e-5f;
    for (int i = 0; i < 256; ++i) tot += red[i];
    msum[t] = tot;
  }
}

__global__ void k_cvt(const float* __restrict__ src, __bf16* __restrict__ dst,
                      int n, int ld, int zdiag) {
  int i = blockIdx.x * 256 + threadIdx.x;
  if (i < n) {
    float v = src[i];
    if (zdiag && (i / ld) == (i % ld)) v = 0.f;   // feat_W * (1 - I)
    dst[i] = (__bf16)v;
  }
}

// ---------------- main persistent RNN kernel ----------------

__global__ __launch_bounds__(256)
void rits_persistent(
    const float* __restrict__ values, const float* __restrict__ masks,
    const float* __restrict__ deltas,
    const float* __restrict__ td_h_b, const float* __restrict__ td_x_W,
    const float* __restrict__ td_x_b, const float* __restrict__ hist_b,
    const float* __restrict__ feat_b, const float* __restrict__ wc_b,
    const float* __restrict__ b_ih,   const float* __restrict__ b_hh,
    const float* __restrict__ out_W,  const float* __restrict__ out_b,
    const __bf16* __restrict__ tdh_g,  const __bf16* __restrict__ hist_g,
    const __bf16* __restrict__ feat_g, const __bf16* __restrict__ wcw_g,
    const __bf16* __restrict__ wih_g,  const __bf16* __restrict__ whh_g,
    const float* __restrict__ msum,
    float* __restrict__ d_loss, float* __restrict__ d_pred, float* __restrict__ d_imp)
{
  __shared__ __align__(16) __bf16 sh_h[2][BM * H_DIM];   // ping-pong hidden state (bf16)
  __shared__ __align__(16) __bf16 sh_dd[BM * D_IN];
  __shared__ __align__(16) __bf16 sh_cat[BM * 2 * D_IN]; // [gamma_x|m] then [c_c|m]
  __shared__ __align__(16) __bf16 sh_xc[BM * D_IN];
  __shared__ float sh_x [BM * D_IN];
  __shared__ float sh_m [BM * D_IN];
  __shared__ float sh_xh[BM * D_IN];
  __shared__ float sh_zh[BM * D_IN];
  __shared__ float sh_red[256];

  const int tid   = threadIdx.x;
  const int wv    = tid >> 5;     // wave 0..7
  const int ln    = tid & 31;
  const int lmod  = ln & 15;      // N within tile / A row
  const int lhalf = ln >> 4;      // K-half select
  const int b0    = blockIdx.x * BM;

  for (int i = tid; i < BM * H_DIM; i += 256) sh_h[0][i] = (__bf16)0.0f;

  float cst[4][8];                // LSTM cell state tiles (wave-owned, in registers)
#pragma unroll
  for (int p = 0; p < 4; ++p)
#pragma unroll
    for (int r = 0; r < 8; ++r) cst[p][r] = 0.f;

  float lsum = 0.f;
  int cur = 0;
  __syncthreads();

  for (int t = 0; t < T_LEN; ++t) {
    __bf16* hc = sh_h[cur];
    __bf16* hn = sh_h[cur ^ 1];
    float lstep = 0.f;

    // Opaque weight pointers: defeat LICM so weight B-fragments are streamed
    // from L2 every iteration instead of being preloaded + spilled to scratch.
    const __bf16* tdh  = opaque_ptr(tdh_g);
    const __bf16* hist = opaque_ptr(hist_g);
    const __bf16* feat = opaque_ptr(feat_g);
    const __bf16* wcw  = opaque_ptr(wcw_g);
    const __bf16* wih  = opaque_ptr(wih_g);
    const __bf16* whh  = opaque_ptr(whh_g);

    // ---- Phase A: load x, m, dd; gamma_x (diag) elementwise; build [gamma_x|m]
    for (int e = tid; e < BM * D_IN; e += 256) {
      int row = e >> 7, col = e & 127;
      size_t gi = (size_t)(b0 + row) * T_LEN * D_IN + (size_t)t * D_IN + col;
      float xv = values[gi], mv = masks[gi], dv = deltas[gi];
      sh_x[e] = xv;
      sh_m[e] = mv;
      sh_dd[e] = (__bf16)dv;
      float gx = __expf(-fmaxf(dv * td_x_W[col * D_IN + col] + td_x_b[col], 0.f));
      sh_cat[row * 256 + col]       = (__bf16)gx;
      sh_cat[row * 256 + 128 + col] = (__bf16)mv;
    }
    __syncthreads();

    // ---- Phase B: gamma_h = exp(-relu(dd @ td_h_W^T + b)); h *= gamma_h (in place)
#pragma unroll
    for (int i = 0; i < 4; ++i) {
      int nt = wv * 4 + i;                       // 32 N-tiles over H=512
      v8f acc = gemm_pipe<4>(sh_dd, 128, tdh, 128, nt * 16 + lmod, lmod, lhalf, zero8());
      int n = nt * 16 + lmod;
      float bia = td_h_b[n];
#pragma unroll
      for (int r = 0; r < 8; ++r) {
        int m = r + lhalf * 8;
        float g  = __expf(-fmaxf(acc[r] + bia, 0.f));
        float hv = (float)hc[m * H_DIM + n];
        hc[m * H_DIM + n] = (__bf16)(hv * g);
      }
    }
    __syncthreads();

    // ---- Phase C: x_h = h @ hist_W^T + hist_b   (one 16x16 N-tile per wave)
    {
      int n = wv * 16 + lmod;
      v8f acc = gemm_pipe<16>(hc, 512, hist, 512, n, lmod, lhalf, zero8());
      float bia = hist_b[n];
#pragma unroll
      for (int r = 0; r < 8; ++r)
        sh_xh[(r + lhalf * 8) * D_IN + n] = acc[r] + bia;
    }
    __syncthreads();

    // ---- Phase C2: loss1, x_c = m*x + (1-m)*x_h
    for (int e = tid; e < BM * D_IN; e += 256) {
      float xv = sh_x[e], mv = sh_m[e], xh = sh_xh[e];
      lstep += fabsf(xv - xh) * mv;
      sh_xc[e] = (__bf16)(mv * xv + (1.f - mv) * xh);
    }
    __syncthreads();

    // ---- Phase D: z_h = x_c @ (feat_W*(1-I))^T + feat_b
    {
      int n = wv * 16 + lmod;
      v8f acc = gemm_pipe<4>(sh_xc, 128, feat, 128, n, lmod, lhalf, zero8());
      float bia = feat_b[n];
#pragma unroll
      for (int r = 0; r < 8; ++r)
        sh_zh[(r + lhalf * 8) * D_IN + n] = acc[r] + bia;
    }
    __syncthreads();

    // ---- Phase E: alpha = [gamma_x|m] @ wc_W^T   (keep in regs across barrier)
    v8f alp;
    {
      int n = wv * 16 + lmod;
      alp = gemm_pipe<8>(sh_cat, 256, wcw, 256, n, lmod, lhalf, zero8());
    }
    __syncthreads();   // all waves done reading sh_cat before c_c overwrites it

    // ---- Phase E2: c_h, c_c, loss2+loss3, imputations, build [c_c|m] in sh_cat
    {
      int n = wv * 16 + lmod;
      float bia = wc_b[n];
#pragma unroll
      for (int r = 0; r < 8; ++r) {
        int m = r + lhalf * 8;
        int e = m * D_IN + n;
        float al = alp[r] + bia;
        float zh = sh_zh[e], xh = sh_xh[e];
        float xv = sh_x[e],  mv = sh_m[e];
        lstep += fabsf(xv - zh) * mv;                 // loss2
        float ch = al * zh + (1.f - al) * xh;
        lstep += fabsf(xv - ch) * mv;                 // loss3
        float cc = mv * xv + (1.f - mv) * ch;
        sh_cat[m * 256 + n] = (__bf16)cc;             // [c_c | m] (m half already there)
        d_imp[(size_t)(b0 + m) * T_LEN * D_IN + (size_t)t * D_IN + n] = cc;
      }
    }
    __syncthreads();

    // ---- Phase F: gates = [c_c|m]@W_ih^T + h@W_hh^T + b ; fused LSTM update
#pragma unroll
    for (int pi = 0; pi < 4; ++pi) {
      int nt = wv * 4 + pi;                        // 32 N-tiles over H=512
      v8f g4[4] = { zero8(), zero8(), zero8(), zero8() };

      FStep curf, nxtf;
      load_fstep(curf, 0, sh_cat, hc, wih, whh, nt, lmod, lhalf);
#pragma unroll
      for (int ks = 0; ks < 24; ++ks) {            // 8 over K=256, 16 over K=512
        if (ks + 1 < 24)
          load_fstep(nxtf, ks + 1, sh_cat, hc, wih, whh, nt, lmod, lhalf);
        g4[0] = WMMA_BF16(g4[0], curf.a, curf.b0);
        g4[1] = WMMA_BF16(g4[1], curf.a, curf.b1);
        g4[2] = WMMA_BF16(g4[2], curf.a, curf.b2);
        g4[3] = WMMA_BF16(g4[3], curf.a, curf.b3);
        curf = nxtf;
      }

      int n = nt * 16 + lmod;
      float bi_i = b_ih[0 * H_DIM + n] + b_hh[0 * H_DIM + n];
      float bi_f = b_ih[1 * H_DIM + n] + b_hh[1 * H_DIM + n];
      float bi_g = b_ih[2 * H_DIM + n] + b_hh[2 * H_DIM + n];
      float bi_o = b_ih[3 * H_DIM + n] + b_hh[3 * H_DIM + n];
#pragma unroll
      for (int r = 0; r < 8; ++r) {
        float ig = fast_sigmoid(g4[0][r] + bi_i);
        float fg = fast_sigmoid(g4[1][r] + bi_f);
        float gg = tanhf(g4[2][r] + bi_g);
        float og = fast_sigmoid(g4[3][r] + bi_o);
        float cn = fg * cst[pi][r] + ig * gg;
        cst[pi][r] = cn;
        int m = r + lhalf * 8;
        hn[m * H_DIM + n] = (__bf16)(og * tanhf(cn));
      }
    }
    __syncthreads();

    // ---- Phase G: predictions y = sigmoid(h_new @ out_W^T + out_b)
    if (tid < BM) {
      float s = out_b[0];
      for (int j = 0; j < H_DIM; ++j) s += (float)hn[tid * H_DIM + j] * out_W[j];
      d_pred[(size_t)(b0 + tid) * T_LEN + t] = fast_sigmoid(s);
    }
    lsum += lstep * (1.0f / msum[t]);
    cur ^= 1;
    // A->B barrier of next iteration protects hn reads in Phase G
  }

  sh_red[tid] = lsum;
  __syncthreads();
  if (tid == 0) {
    float tot = 0.f;
    for (int i = 0; i < 256; ++i) tot += sh_red[i];
    atomicAdd(d_loss, tot * (1.0f / (float)T_LEN));
  }
}

// ---------------- host launcher ----------------

// ws layout (bytes)
#define OFF_MSUM 0u
#define OFF_TDH  1024u                          // 65536 bf16
#define OFF_HIST (OFF_TDH  + 131072u)           // 65536 bf16
#define OFF_FEAT (OFF_HIST + 131072u)           // 16384 bf16
#define OFF_WC   (OFF_FEAT + 32768u)            // 32768 bf16
#define OFF_WIH  (OFF_WC   + 65536u)            // 524288 bf16
#define OFF_WHH  (OFF_WIH  + 1048576u)          // 1048576 bf16  -> end ~3.35 MB

extern "C" void kernel_launch(void* const* d_in, const int* in_sizes, int n_in,
                              void* d_out, int out_size, void* d_ws, size_t ws_size,
                              hipStream_t stream) {
  (void)in_sizes; (void)n_in; (void)out_size; (void)ws_size;

  const float* values = (const float*)d_in[0];
  const float* masks  = (const float*)d_in[1];
  const float* deltas = (const float*)d_in[2];
  const float* td_h_W = (const float*)d_in[3];
  const float* td_h_b = (const float*)d_in[4];
  const float* td_x_W = (const float*)d_in[5];
  const float* td_x_b = (const float*)d_in[6];
  const float* hist_W = (const float*)d_in[7];
  const float* hist_b = (const float*)d_in[8];
  const float* feat_W = (const float*)d_in[9];
  const float* feat_b = (const float*)d_in[10];
  const float* wc_W   = (const float*)d_in[11];
  const float* wc_b   = (const float*)d_in[12];
  const float* W_ih   = (const float*)d_in[13];
  const float* b_ih   = (const float*)d_in[14];
  const float* W_hh   = (const float*)d_in[15];
  const float* b_hh   = (const float*)d_in[16];
  const float* out_W  = (const float*)d_in[17];
  const float* out_b  = (const float*)d_in[18];

  char* ws = (char*)d_ws;
  float*  msum = (float*)(ws + OFF_MSUM);
  __bf16* tdh  = (__bf16*)(ws + OFF_TDH);
  __bf16* hist = (__bf16*)(ws + OFF_HIST);
  __bf16* feat = (__bf16*)(ws + OFF_FEAT);
  __bf16* wcw  = (__bf16*)(ws + OFF_WC);
  __bf16* wih  = (__bf16*)(ws + OFF_WIH);
  __bf16* whh  = (__bf16*)(ws + OFF_WHH);

  float* d_loss = (float*)d_out;
  float* d_pred = d_loss + 1;
  float* d_imp  = d_pred + (size_t)B_TOT * T_LEN;

  k_zero_loss<<<1, 64, 0, stream>>>(d_loss);
  k_msum<<<T_LEN, 256, 0, stream>>>(masks, msum);

  k_cvt<<<(65536   + 255) / 256, 256, 0, stream>>>(td_h_W, tdh,  65536,  128, 0);
  k_cvt<<<(65536   + 255) / 256, 256, 0, stream>>>(hist_W, hist, 65536,  512, 0);
  k_cvt<<<(16384   + 255) / 256, 256, 0, stream>>>(feat_W, feat, 16384,  128, 1);
  k_cvt<<<(32768   + 255) / 256, 256, 0, stream>>>(wc_W,   wcw,  32768,  256, 0);
  k_cvt<<<(524288  + 255) / 256, 256, 0, stream>>>(W_ih,   wih,  524288, 256, 0);
  k_cvt<<<(1048576 + 255) / 256, 256, 0, stream>>>(W_hh,   whh,  1048576,512, 0);

  rits_persistent<<<B_TOT / BM, 256, 0, stream>>>(
      values, masks, deltas,
      td_h_b, td_x_W, td_x_b, hist_b, feat_b, wc_b, b_ih, b_hh, out_W, out_b,
      tdh, hist, feat, wcw, wih, whh, msum,
      d_loss, d_pred, d_imp);
}
